// trajectory2seq_25168508355311
// MI455X (gfx1250) — matrix operand order repaired and checked by reading, not compiled, not run on hardware
//
#include <hip/hip_runtime.h>

// ---------------- types ----------------
typedef __bf16 bf16;
typedef __attribute__((ext_vector_type(16))) __bf16 v16bf;
typedef __attribute__((ext_vector_type(8)))  float  v8f;

#define Bsz 128
#define Ssz 512
#define Hsz 512
#define Vsz 30
#define Tsz 64
#define Gsz 1536   // 3*H

// WMMA fragment helpers.
// A-fragment (16x32 bf16): lane m = lane&15, half = lane>>4.
//   elements 0..7  hold K = k0 + 16*half + (0..7)
//   elements 8..15 hold K = k0 + 16*half + 8 + (0..7)
template <typename AT>
__device__ __forceinline__ v16bf load_afrag(const AT* __restrict__ Arow, int k0, int half) {
    v16bf af;
    const AT* p0 = Arow + k0 + 16 * half;
    const AT* p1 = p0 + 8;
#pragma unroll
    for (int e = 0; e < 8; ++e) {
        af[e]     = (bf16)p0[e];
        af[8 + e] = (bf16)p1[e];
    }
    return af;
}
// B-fragment (32x16 bf16): lane n = lane&15, lanes 16..31 take K 16..31; 16 contiguous K.
__device__ __forceinline__ v16bf load_bfrag(const bf16* __restrict__ W, int col, int m,
                                            int k0, int half) {
    return *(const v16bf*)(W + (size_t)(col + m) * Hsz + k0 + half * 16);
}
__device__ __forceinline__ v8f wmma_bf16(v16bf a, v16bf b, v8f c) {
    return __builtin_amdgcn_wmma_f32_16x16x32_bf16(false, a, false, b, (short)0, c, false, false);
}
__device__ __forceinline__ float sigmoidf_(float x) { return 1.0f / (1.0f + __expf(-x)); }

// ---------------- f32 -> bf16 weight conversion ----------------
__global__ void f32_to_bf16_k(const float* __restrict__ src, bf16* __restrict__ dst, int n) {
    int i = blockIdx.x * blockDim.x + threadIdx.x;
    if (i < n) dst[i] = (bf16)src[i];
}

// ================= fused GRU cell (general): gi & gh GEMMs + gates, all in registers =========
// One wave owns a 16-row x 32-h-col patch across ALL THREE gate blocks of both GEMMs,
// so r/z/n math happens on register accumulators with identical C layouts. No LDS, no scratch.
// Grid: (Bsz/16, Hsz/32) single-wave blocks.  hOld -> hNew (ping-pong, no in-place race).
template <typename AT, bool EMB>
__global__ __launch_bounds__(32) void gru_cell_k(
    const AT* __restrict__ Ain,          // activations [128 x 512], or emb table [V x 512] if EMB
    const int* __restrict__ tok,         // token ids (EMB only)
    const float* __restrict__ hOld,      // [128 x 512]
    const bf16* __restrict__ Wih,        // [1536 x 512] bf16 row-major
    const bf16* __restrict__ Whh,        // [1536 x 512]
    const float* __restrict__ bih, const float* __restrict__ bhh,
    float* __restrict__ hNew,            // [128 x 512]
    bf16* __restrict__ seq, long long rstride) {
    const int lane    = threadIdx.x & 31;
    const int mTile   = blockIdx.x * 16;
    const int colBase = blockIdx.y * 32;        // h-column group (32 wide)
    const int m       = lane & 15;
    const int half    = lane >> 4;

    v8f accI[3][2] = {};
    v8f accH[3][2] = {};

    const int arow = mTile + m;
    const AT* AinRow = EMB ? (Ain + (size_t)tok[arow] * Hsz) : (Ain + (size_t)arow * Hsz);
    const float* HRow = hOld + (size_t)arow * Hsz;

    for (int k0 = 0; k0 < Hsz; k0 += 32) {
        v16bf aI = load_afrag<AT>(AinRow, k0, half);
        v16bf aH = load_afrag<float>(HRow, k0, half);
#pragma unroll
        for (int g = 0; g < 3; ++g) {
#pragma unroll
            for (int jt = 0; jt < 2; ++jt) {
                const int col = g * Hsz + colBase + jt * 16;
                accI[g][jt] = wmma_bf16(aI, load_bfrag(Wih, col, m, k0, half), accI[g][jt]);
                accH[g][jt] = wmma_bf16(aH, load_bfrag(Whh, col, m, k0, half), accH[g][jt]);
            }
        }
    }

    // gate math, entirely in registers (all six accumulators share the C layout)
#pragma unroll
    for (int jt = 0; jt < 2; ++jt) {
        const int col = colBase + jt * 16 + m;
        const float bir = bih[col], biz = bih[Hsz + col], bin = bih[2 * Hsz + col];
        const float bhr = bhh[col], bhz = bhh[Hsz + col], bhn = bhh[2 * Hsz + col];
#pragma unroll
        for (int r = 0; r < 8; ++r) {
            const int row = mTile + r + 8 * half;
            float ir = accI[0][jt][r] + bir, iz = accI[1][jt][r] + biz, in_ = accI[2][jt][r] + bin;
            float hr = accH[0][jt][r] + bhr, hz = accH[1][jt][r] + bhz, hn = accH[2][jt][r] + bhn;
            float rg = sigmoidf_(ir + hr);
            float zg = sigmoidf_(iz + hz);
            float ng = tanhf(in_ + rg * hn);
            float hv = hOld[(size_t)row * Hsz + col];
            float hnew = (1.0f - zg) * ng + zg * hv;
            hNew[(size_t)row * Hsz + col] = hnew;
            if (seq) seq[(size_t)row * rstride + col] = (bf16)hnew;
        }
    }
}

// ================= fused GRU cell, encoder layer 0 (input dim = 2, gi elementwise) ==========
__global__ __launch_bounds__(32) void gru_cell_l0_k(
    const float* __restrict__ x,          // [B, S, 2]
    const float* __restrict__ Wih0,       // [1536, 2] f32
    const float* __restrict__ bih0,
    const float* __restrict__ hOld, const bf16* __restrict__ Whh,
    const float* __restrict__ bhh,
    float* __restrict__ hNew, bf16* __restrict__ seq, long long rstride, int t) {
    const int lane    = threadIdx.x & 31;
    const int mTile   = blockIdx.x * 16;
    const int colBase = blockIdx.y * 32;
    const int m       = lane & 15;
    const int half    = lane >> 4;

    v8f accH[3][2] = {};
    const float* HRow = hOld + (size_t)(mTile + m) * Hsz;

    for (int k0 = 0; k0 < Hsz; k0 += 32) {
        v16bf aH = load_afrag<float>(HRow, k0, half);
#pragma unroll
        for (int g = 0; g < 3; ++g) {
#pragma unroll
            for (int jt = 0; jt < 2; ++jt) {
                const int col = g * Hsz + colBase + jt * 16;
                accH[g][jt] = wmma_bf16(aH, load_bfrag(Whh, col, m, k0, half), accH[g][jt]);
            }
        }
    }

#pragma unroll
    for (int jt = 0; jt < 2; ++jt) {
        const int col = colBase + jt * 16 + m;
        float wr0 = Wih0[col * 2], wr1 = Wih0[col * 2 + 1];
        float wz0 = Wih0[(Hsz + col) * 2], wz1 = Wih0[(Hsz + col) * 2 + 1];
        float wn0 = Wih0[(2 * Hsz + col) * 2], wn1 = Wih0[(2 * Hsz + col) * 2 + 1];
        const float bir = bih0[col], biz = bih0[Hsz + col], bin = bih0[2 * Hsz + col];
        const float bhr = bhh[col],  bhz = bhh[Hsz + col],  bhn = bhh[2 * Hsz + col];
#pragma unroll
        for (int r = 0; r < 8; ++r) {
            const int row = mTile + r + 8 * half;
            const float x0 = x[((size_t)row * Ssz + t) * 2 + 0];
            const float x1 = x[((size_t)row * Ssz + t) * 2 + 1];
            float ir = x0 * wr0 + x1 * wr1 + bir;
            float iz = x0 * wz0 + x1 * wz1 + biz;
            float in_ = x0 * wn0 + x1 * wn1 + bin;
            float hr = accH[0][jt][r] + bhr, hz = accH[1][jt][r] + bhz, hn = accH[2][jt][r] + bhn;
            float rg = sigmoidf_(ir + hr);
            float zg = sigmoidf_(iz + hz);
            float ng = tanhf(in_ + rg * hn);
            float hv = hOld[(size_t)row * Hsz + col];
            float hnew = (1.0f - zg) * ng + zg * hv;
            hNew[(size_t)row * Hsz + col] = hnew;
            seq[(size_t)row * rstride + col] = (bf16)hnew;
        }
    }
}

// ---------------- attention: sim[b,s] = <out_enc[b,s,:], d1[b,:]> ----------------
__global__ void attn_sim_k(const bf16* __restrict__ out_enc, const float* __restrict__ d1,
                           float* __restrict__ sim) {
    int i = blockIdx.x * blockDim.x + threadIdx.x;
    if (i >= Bsz * Ssz) return;
    int b = i >> 9, s = i & (Ssz - 1);
    const bf16* row = out_enc + ((size_t)b * Ssz + s) * Hsz;
    const float* d  = d1 + (size_t)b * Hsz;
    float acc = 0.0f;
#pragma unroll 4
    for (int k = 0; k < Hsz; ++k) acc += (float)row[k] * d[k];
    sim[i] = acc;
}

// ---------------- softmax over BATCH axis (faithful to reference axis=0), in-place ----------
__global__ void softmax_batch_k(float* __restrict__ sim) {
    int s = blockIdx.x;          // 512 columns
    int b = threadIdx.x;         // 128 batch
    __shared__ float red[Bsz];
    float v = sim[(size_t)b * Ssz + s];
    red[b] = v; __syncthreads();
    for (int off = 64; off > 0; off >>= 1) {
        if (b < off) red[b] = fmaxf(red[b], red[b + off]);
        __syncthreads();
    }
    float mx = red[0]; __syncthreads();
    float ev = __expf(v - mx);
    red[b] = ev; __syncthreads();
    for (int off = 64; off > 0; off >>= 1) {
        if (b < off) red[b] += red[b + off];
        __syncthreads();
    }
    sim[(size_t)b * Ssz + s] = ev / red[0];
}

// ---------------- attention context: attn[b,h] = sum_s w[b,s]*out_enc[b,s,h] ----------------
__global__ void attn_ctx_k(const float* __restrict__ attw, const bf16* __restrict__ out_enc,
                           float* __restrict__ attn) {
    int i = blockIdx.x * blockDim.x + threadIdx.x;
    if (i >= Bsz * Hsz) return;
    int b = i >> 9, hh = i & (Hsz - 1);
    const float* w   = attw + (size_t)b * Ssz;
    const bf16*  enc = out_enc + (size_t)b * Ssz * Hsz + hh;
    float acc = 0.0f;
#pragma unroll 4
    for (int s = 0; s < Ssz; ++s) acc += w[s] * (float)enc[(size_t)s * Hsz];
    attn[i] = acc;
}

// ---------------- fc logits + argmax feedback ----------------
__global__ __launch_bounds__(32) void fc_argmax_k(const float* __restrict__ d1,
                                                  const float* __restrict__ attn,
                                                  const float* __restrict__ fcW,
                                                  const float* __restrict__ fcb,
                                                  float* __restrict__ out, int* __restrict__ tok,
                                                  int t) {
    int b = blockIdx.x;
    int v = threadIdx.x;
    float logit = -1e30f;
    if (v < Vsz) {
        const float* wv = fcW + (size_t)v * (2 * Hsz);
        float acc = fcb[v];
        const float* d = d1 + (size_t)b * Hsz;
        const float* a = attn + (size_t)b * Hsz;
#pragma unroll 4
        for (int k = 0; k < Hsz; ++k) acc += d[k] * wv[k];
#pragma unroll 4
        for (int k = 0; k < Hsz; ++k) acc += a[k] * wv[Hsz + k];
        logit = acc;
        out[(size_t)b * Tsz * Vsz + (size_t)t * Vsz + v] = acc;
    }
    float best = logit; int bi = v;
    for (int off = 16; off > 0; off >>= 1) {
        float ov = __shfl_down(best, off, 32);
        int   oi = __shfl_down(bi, off, 32);
        if (ov > best || (ov == best && oi < bi)) { best = ov; bi = oi; }
    }
    if (v == 0) tok[b] = bi;
}

// ---------------- final hidden copy ----------------
__global__ void copy_hidden_k(const float* __restrict__ h0, const float* __restrict__ h1,
                              float* __restrict__ out) {
    int i = blockIdx.x * blockDim.x + threadIdx.x;
    if (i >= Bsz * Hsz) return;
    out[i] = h0[i];
    out[Bsz * Hsz + i] = h1[i];
}

// ---------------- host orchestration ----------------
extern "C" void kernel_launch(void* const* d_in, const int* in_sizes, int n_in,
                              void* d_out, int out_size, void* d_ws, size_t ws_size,
                              hipStream_t stream) {
    const float* x        = (const float*)d_in[0];
    const float* emb      = (const float*)d_in[1];
    const float* enc_Wih0 = (const float*)d_in[2];
    const float* enc_Whh0 = (const float*)d_in[3];
    const float* enc_bih0 = (const float*)d_in[4];
    const float* enc_bhh0 = (const float*)d_in[5];
    const float* enc_Wih1 = (const float*)d_in[6];
    const float* enc_Whh1 = (const float*)d_in[7];
    const float* enc_bih1 = (const float*)d_in[8];
    const float* enc_bhh1 = (const float*)d_in[9];
    const float* dec_Wih0 = (const float*)d_in[10];
    const float* dec_Whh0 = (const float*)d_in[11];
    const float* dec_bih0 = (const float*)d_in[12];
    const float* dec_bhh0 = (const float*)d_in[13];
    const float* dec_Wih1 = (const float*)d_in[14];
    const float* dec_Whh1 = (const float*)d_in[15];
    const float* dec_bih1 = (const float*)d_in[16];
    const float* dec_bhh1 = (const float*)d_in[17];
    const float* fc_W     = (const float*)d_in[18];
    const float* fc_b     = (const float*)d_in[19];
    float* out = (float*)d_out;

    // ----- workspace carve-up -----
    char* ws = (char*)d_ws;
    size_t off = 0;
    auto carve = [&](size_t bytes) { char* p = ws + off; off = (off + bytes + 255) & ~(size_t)255; return p; };
    const size_t WK = (size_t)Gsz * Hsz;
    bf16* wWhh0e = (bf16*)carve(WK * sizeof(bf16));
    bf16* wWih1e = (bf16*)carve(WK * sizeof(bf16));
    bf16* wWhh1e = (bf16*)carve(WK * sizeof(bf16));
    bf16* wWih0d = (bf16*)carve(WK * sizeof(bf16));
    bf16* wWhh0d = (bf16*)carve(WK * sizeof(bf16));
    bf16* wWih1d = (bf16*)carve(WK * sizeof(bf16));
    bf16* wWhh1d = (bf16*)carve(WK * sizeof(bf16));
    bf16* a_seq   = (bf16*)carve((size_t)Ssz * Bsz * Hsz * sizeof(bf16));  // [t][b][h]
    bf16* out_enc = (bf16*)carve((size_t)Bsz * Ssz * Hsz * sizeof(bf16));  // [b][s][h]
    float* h0a = (float*)carve((size_t)Bsz * Hsz * sizeof(float));
    float* h0b = (float*)carve((size_t)Bsz * Hsz * sizeof(float));
    float* h1a = (float*)carve((size_t)Bsz * Hsz * sizeof(float));
    float* h1b = (float*)carve((size_t)Bsz * Hsz * sizeof(float));
    float* sim  = (float*)carve((size_t)Bsz * Ssz * sizeof(float));
    float* attn = (float*)carve((size_t)Bsz * Hsz * sizeof(float));
    int*   tok  = (int*)carve((size_t)Bsz * sizeof(int));
    (void)ws_size; (void)n_in; (void)in_sizes; (void)out_size;

    const int T256 = 256;
    const dim3 cellGrid(Bsz / 16, Hsz / 32);   // 8 x 16 single-wave blocks
    const int nBH = (Bsz * Hsz + T256 - 1) / T256;
    const int nBS = (Bsz * Ssz + T256 - 1) / T256;
    const int nWK = (int)((WK + T256 - 1) / T256);

    // ----- per-call prep: weight conversion + state init -----
    f32_to_bf16_k<<<nWK, T256, 0, stream>>>(enc_Whh0, wWhh0e, (int)WK);
    f32_to_bf16_k<<<nWK, T256, 0, stream>>>(enc_Wih1, wWih1e, (int)WK);
    f32_to_bf16_k<<<nWK, T256, 0, stream>>>(enc_Whh1, wWhh1e, (int)WK);
    f32_to_bf16_k<<<nWK, T256, 0, stream>>>(dec_Wih0, wWih0d, (int)WK);
    f32_to_bf16_k<<<nWK, T256, 0, stream>>>(dec_Whh0, wWhh0d, (int)WK);
    f32_to_bf16_k<<<nWK, T256, 0, stream>>>(dec_Wih1, wWih1d, (int)WK);
    f32_to_bf16_k<<<nWK, T256, 0, stream>>>(dec_Whh1, wWhh1d, (int)WK);
    hipMemsetAsync(h0a, 0, (size_t)Bsz * Hsz * sizeof(float), stream);
    hipMemsetAsync(h1a, 0, (size_t)Bsz * Hsz * sizeof(float), stream);
    hipMemsetAsync(tok, 0, (size_t)Bsz * sizeof(int), stream);

    float *h0cur = h0a, *h0nxt = h0b, *h1cur = h1a, *h1nxt = h1b;
    auto swap0 = [&]() { float* t_ = h0cur; h0cur = h0nxt; h0nxt = t_; };
    auto swap1 = [&]() { float* t_ = h1cur; h1cur = h1nxt; h1nxt = t_; };

    // ----- encoder: 512 sequential steps, 2 fused launches per step -----
    for (int t = 0; t < Ssz; ++t) {
        gru_cell_l0_k<<<cellGrid, 32, 0, stream>>>(
            x, enc_Wih0, enc_bih0, h0cur, wWhh0e, enc_bhh0, h0nxt,
            a_seq + (size_t)t * Bsz * Hsz, (long long)Hsz, t);
        swap0();
        gru_cell_k<bf16, false><<<cellGrid, 32, 0, stream>>>(
            a_seq + (size_t)t * Bsz * Hsz, (const int*)nullptr, h1cur,
            wWih1e, wWhh1e, enc_bih1, enc_bhh1, h1nxt,
            out_enc + (size_t)t * Hsz, (long long)(Ssz * Hsz));
        swap1();
    }

    // ----- decoder: 64 autoregressive steps with attention -----
    for (int t = 0; t < Tsz; ++t) {
        gru_cell_k<float, true><<<cellGrid, 32, 0, stream>>>(
            emb, tok, h0cur, wWih0d, wWhh0d, dec_bih0, dec_bhh0, h0nxt,
            (bf16*)nullptr, 0);
        swap0();
        gru_cell_k<float, false><<<cellGrid, 32, 0, stream>>>(
            h0cur, (const int*)nullptr, h1cur, wWih1d, wWhh1d, dec_bih1, dec_bhh1, h1nxt,
            (bf16*)nullptr, 0);
        swap1();
        attn_sim_k<<<nBS, T256, 0, stream>>>(out_enc, h1cur, sim);
        softmax_batch_k<<<Ssz, Bsz, 0, stream>>>(sim);
        attn_ctx_k<<<nBH, T256, 0, stream>>>(sim, out_enc, attn);
        fc_argmax_k<<<Bsz, 32, 0, stream>>>(h1cur, attn, fc_W, fc_b, out, tok, t);
    }

    // ----- final hidden: [2, B, H] appended after logits -----
    copy_hidden_k<<<nBH, T256, 0, stream>>>(h0cur, h1cur, out + (size_t)Bsz * Tsz * Vsz);
}
